// SpectrumMambaS4_35459249995881
// MI455X (gfx1250) — matrix operand order repaired and checked by reading, not compiled, not run on hardware
//
#include <hip/hip_runtime.h>
#include <hip/hip_bf16.h>

// SpectrumMambaS4 on gfx1250: f16 WMMA GEMMs (f32 accumulate), fused LN
// epilogues, chunk-parallel linear scan, TDM (tensor_load_to_lds) staging of
// f16 A-tiles with hardware LDS padding, global prefetch of weight streams.
// Staging loops are split load-then-convert to keep 8 global_load_b128 in
// flight per thread instead of load/wait/store lockstep.

typedef __attribute__((ext_vector_type(16))) _Float16 v16h;
typedef __attribute__((ext_vector_type(8)))  float    v8f;
typedef unsigned int __attribute__((ext_vector_type(4))) v4u;
typedef int __attribute__((ext_vector_type(8))) v8i;
typedef int __attribute__((ext_vector_type(4))) v4i;

#define TT    2048
#define DD    256
#define SS    64
#define BB    64
#define MT    131072   // BB*TT
#define NCHK  32       // chunks of 64 along T
#define CLEN  64       // chunk length

// ---------------- WMMA fragment helpers (wave32, per ISA 7.12.2) ----------

// A-matrix 16x32 f16: lanes 0-15 rows M, K-octet base 0; lanes 16-31 same M,
// K-octet base 8. halfs 0..7 -> K kb..kb+7, halfs 8..15 -> K kb+16..kb+23.
__device__ __forceinline__ v16h frag_a(const _Float16* As, int lda, int row0, int k0) {
  const int lane = threadIdx.x & 31;
  const _Float16* p = As + (size_t)(row0 + (lane & 15)) * lda + k0 + ((lane >> 4) << 3);
  v16h a;
#pragma unroll
  for (int i = 0; i < 8; ++i) { a[i] = p[i]; a[i + 8] = p[i + 16]; }
  return a;
}

// B-matrix 32x16 f16 stored in LDS as [n][k]: lane = N col, halfs = K
// (lanes 0-15: K 0..7,16..23, lanes 16-31: K 8..15,24..31).
__device__ __forceinline__ v16h frag_b(const _Float16* Bs, int ldb, int n0, int k0) {
  const int lane = threadIdx.x & 31;
  const _Float16* p = Bs + (size_t)(n0 + (lane & 15)) * ldb + k0 + ((lane >> 4) << 3);
  v16h b;
#pragma unroll
  for (int i = 0; i < 8; ++i) { b[i] = p[i]; b[i + 8] = p[i + 16]; }
  return b;
}

__device__ __forceinline__ v8f wmma_f16(v16h a, v16h b, v8f c) {
  return __builtin_amdgcn_wmma_f32_16x16x32_f16(false, a, false, b, (short)0, c, false, false);
}

__device__ __forceinline__ float wave_sum32(float v) {
#pragma unroll
  for (int off = 16; off > 0; off >>= 1) v += __shfl_xor(v, off, 32);
  return v;
}

__device__ __forceinline__ void st4h(_Float16* d, float4 v) {
  d[0] = (_Float16)v.x; d[1] = (_Float16)v.y; d[2] = (_Float16)v.z; d[3] = (_Float16)v.w;
}

__device__ __forceinline__ void zero_acc(v8f* a) {
#pragma unroll
  for (int j = 0; j < 4; ++j)
#pragma unroll
    for (int r = 0; r < 8; ++r) a[j][r] = 0.f;
}

// Stage NJ*256 float4s (NJ per thread) from a row-major f32 panel (row stride
// `gs` floats, 32-float k-window at kc) into f16 LDS with 40-half pitch.
// Loads are batched before converts to keep them all in flight.
template <int NJ>
__device__ __forceinline__ void stage_b_f32(const float* __restrict__ g, int gs,
                                            int kc, int klimit, _Float16* Ls) {
  float4 vals[NJ];
#pragma unroll
  for (int j = 0; j < NJ; ++j) {
    int idx = (threadIdx.x + j * 256) * 4;
    int r = idx >> 5, k = idx & 31;
    vals[j] = *(const float4*)(g + (size_t)r * gs + kc + k);
  }
#pragma unroll
  for (int j = 0; j < NJ; ++j) {
    int idx = (threadIdx.x + j * 256) * 4;
    int r = idx >> 5, k = idx & 31;
    st4h(Ls + r * 40 + k, vals[j]);
    if (kc + 32 < klimit)
      __builtin_prefetch(g + (size_t)r * gs + kc + 32 + k, 0, 3);
  }
}

// ---------------- TDM: 2D f16 tile global->LDS (ISA 8.3/8.4) --------------
// Loads tile_h rows x 32 halfs from a row-major f16 tensor (row stride
// `stride_elems` halfs) starting at gptr, into LDS at lds_off with hardware
// padding: pad_interval=3 (every 16 DWORDs = 32 halfs) + pad_amount=3
// (4 DWORDs = 8 halfs) -> LDS row pitch 40 halfs, matching frag_a layout.
// Issue from one wave; completion via TENSORcnt. This toolchain exposes the
// 6-arg builtin: (v4u g0, v8i g1, v4i g2, v4i g3, v8i extra, i32 cpol).
__device__ __forceinline__ void tdm_load_a32(const _Float16* gptr, unsigned lds_off,
                                             unsigned tile_h, unsigned stride_elems) {
  unsigned long long ga = (unsigned long long)(uintptr_t)(const void*)gptr;
  v4u g0;
  g0[0] = 1u;                                   // count=1 (valid), user mode
  g0[1] = lds_off;                              // lds_addr (bytes)
  g0[2] = (unsigned)(ga & 0xFFFFFFFFu);         // global_addr[31:0]
  g0[3] = (unsigned)((ga >> 32) & 0x1FFFFFFu)   // global_addr[56:32]
          | (2u << 30);                         // type=2 ("image")
  v8i g1;
  g1[0] = (int)((1u << 16)      // data_size=1 -> 2-byte elements
              | (1u << 20)      // pad_enable
              | (3u << 22)      // pad_interval: 16 DWORDs
              | (3u << 25));    // pad_amount:   4 DWORDs
  g1[1] = (int)((stride_elems & 0xFFFFu) << 16);          // tensor_dim0[15:0]
  g1[2] = (int)((stride_elems >> 16) |
                ((tile_h & 0xFFFFu) << 16));              // dim0[31:16] | dim1[15:0]
  g1[3] = (int)((tile_h >> 16) | (32u << 16));            // dim1[31:16] | tile_dim0=32
  g1[4] = (int)(tile_h & 0xFFFFu);                        // tile_dim1 (tile_dim2=0)
  g1[5] = (int)stride_elems;                              // tensor_dim0_stride[31:0]
  g1[6] = 0;                                              // stride[47:32], dim1_stride lo
  g1[7] = 0;
  v4i z4 = {0, 0, 0, 0};                                  // 2D: groups 2/3 zero
  v8i z8 = {0, 0, 0, 0, 0, 0, 0, 0};
  __builtin_amdgcn_tensor_load_to_lds(g0, g1, z4, z4, z8, 0);
}

__device__ __forceinline__ unsigned lds_offset_of(const void* p) {
  return (unsigned)(uintptr_t)p;  // low 32 bits of flat shared addr = LDS offset
}

// ---------------- embed: h[b,t,d] = x[b,t]*emb_w[d] + emb_b[d] ------------

__global__ __launch_bounds__(256) void k_embed(const float* __restrict__ x,
                                               const float* __restrict__ ew,
                                               const float* __restrict__ eb,
                                               float* __restrict__ h) {
  size_t i = (size_t)blockIdx.x * 256 + threadIdx.x;
  int d = (int)(i & 255);
  size_t bt = i >> 8;
  h[i] = x[bt] * ew[d] + eb[d];
}

// ---------------- in_proj + bias + LayerNorm -> u (f16) -------------------
// C[m,n] = sum_k h[m,k]*in_w[n,k]; block tile 32(M) x 256(N), 8 waves.

__global__ __launch_bounds__(256) void k_inproj_ln(
    const float* __restrict__ h, const float* __restrict__ Wg,
    const float* __restrict__ bg, const float* __restrict__ lwg,
    const float* __restrict__ lbg, _Float16* __restrict__ u16, int l) {
  __shared__ union __align__(16) {
    struct { _Float16 As[32 * 40]; _Float16 Bs[256 * 40]; } g;
    float yb[32 * 264];
  } sm;
  __shared__ float bias[256], gw[256], gb[256];

  const float* W = Wg + (size_t)l * 65536;
  {
    int t = threadIdx.x;
    bias[t] = bg[l * 256 + t];
    gw[t]   = lwg[l * 256 + t];
    gb[t]   = lbg[l * 256 + t];
  }
  const int rb = blockIdx.x * 32;
  const int lane = threadIdx.x & 31, wave = threadIdx.x >> 5;
  const int wrow = (wave >> 2) * 16, wcol = (wave & 3) * 64;

  v8f acc[4];
  zero_acc(acc);

  for (int kc = 0; kc < 256; kc += 32) {
    __syncthreads();
    {  // stage A: 32x32 f32 -> f16 (conversion: keep on VALU)
      int idx = threadIdx.x * 4;
      int r = idx >> 5, k = idx & 31;
      float4 v = *(const float4*)(h + (size_t)(rb + r) * 256 + kc + k);
      st4h(sm.g.As + r * 40 + k, v);
    }
    stage_b_f32<8>(W, 256, kc, 256, sm.g.Bs);  // 256(n) x 32(k) from in_w[n][k]
    __syncthreads();
    v16h af = frag_a(sm.g.As, 40, wrow, 0);
#pragma unroll
    for (int j = 0; j < 4; ++j) {
      v16h bf = frag_b(sm.g.Bs, 40, wcol + j * 16, 0);
      acc[j] = wmma_f16(af, bf, acc[j]);
    }
  }
  __syncthreads();
  {  // write accumulators (+bias) to LDS (overlaps tiles; protected by barrier)
    const int rh = lane >> 4, col = lane & 15;
#pragma unroll
    for (int j = 0; j < 4; ++j) {
      int c = wcol + j * 16 + col;
      float bv = bias[c];
#pragma unroll
      for (int r = 0; r < 8; ++r)
        sm.yb[(wrow + rh * 8 + r) * 264 + c] = acc[j][r] + bv;
    }
  }
  __syncthreads();
#pragma unroll
  for (int rr = 0; rr < 4; ++rr) {  // LayerNorm, one wave per 4 rows
    int row = wave * 4 + rr;
    const float* yr = sm.yb + row * 264;
    int c0 = lane * 8;
    float v0[8];
#pragma unroll
    for (int i = 0; i < 8; ++i) v0[i] = yr[c0 + i];
    float s = 0.f;
#pragma unroll
    for (int i = 0; i < 8; ++i) s += v0[i];
    float mean = wave_sum32(s) * (1.f / 256.f);
    float q = 0.f;
#pragma unroll
    for (int i = 0; i < 8; ++i) { float d = v0[i] - mean; q += d * d; }
    float rs = rsqrtf(wave_sum32(q) * (1.f / 256.f) + 1e-5f);
    _Float16* dst = u16 + (size_t)(rb + row) * 256 + c0;
#pragma unroll
    for (int i = 0; i < 8; ++i)
      dst[i] = (_Float16)((v0[i] - mean) * rs * gw[c0 + i] + gb[c0 + i]);
  }
}

// ---------------- uB = u @ Bm^T : M x 64, K=256 ---------------------------

__global__ __launch_bounds__(256) void k_uB(const _Float16* __restrict__ u16,
                                            const float* __restrict__ Bmg,
                                            float* __restrict__ uB, int l) {
  __shared__ __align__(16) _Float16 As[128 * 40];
  __shared__ __align__(16) _Float16 Bs[64 * 40];
  const float* W = Bmg + (size_t)l * 64 * 256;
  const int rb = blockIdx.x * 128;
  const int lane = threadIdx.x & 31, wave = threadIdx.x >> 5;
  const int wrow = wave * 16;
  const unsigned as_off = lds_offset_of(As);
  v8f acc[4];
  zero_acc(acc);

  for (int kc = 0; kc < 256; kc += 32) {
    __syncthreads();
    // stage A via TDM: 128 rows x 32 halfs of u16, HW-padded to 40-half pitch
    if (threadIdx.x < 32)
      tdm_load_a32(u16 + (size_t)rb * 256 + kc, as_off, 128, 256);
    stage_b_f32<2>(W, 256, kc, 256, Bs);  // 64(n) x 32(k) from Bm[n][k]
    if (threadIdx.x < 32) __builtin_amdgcn_s_wait_tensorcnt(0);
    __syncthreads();
    v16h af = frag_a(As, 40, wrow, 0);
#pragma unroll
    for (int j = 0; j < 4; ++j) {
      v16h bf = frag_b(Bs, 40, j * 16, 0);
      acc[j] = wmma_f16(af, bf, acc[j]);
    }
  }
  const int rh = lane >> 4, col = lane & 15;
#pragma unroll
  for (int j = 0; j < 4; ++j)
#pragma unroll
    for (int r = 0; r < 8; ++r)
      uB[(size_t)(rb + wrow + rh * 8 + r) * 64 + j * 16 + col] = acc[j][r];
}

// ---------------- scan: P[k] = A^k, k=1..64 (one block) -------------------

__global__ __launch_bounds__(256) void s0_powers(const float* __restrict__ Am,
                                                 float* __restrict__ P, int l) {
  __shared__ float Aij[4096];
  __shared__ float Pp[4096];
  const float* A = Am + (size_t)l * 4096;
  for (int i = threadIdx.x; i < 4096; i += 256) {
    float v = A[i];
    Aij[i] = v; Pp[i] = v; P[i] = v;  // P[1] = A at offset 0
  }
  __syncthreads();
  for (int k = 2; k <= CLEN; ++k) {
    float out[16];
#pragma unroll
    for (int j = 0; j < 16; ++j) {
      int o = threadIdx.x + j * 256;
      int i = o >> 6, c = o & 63;
      float acc = 0.f;
#pragma unroll 8
      for (int m = 0; m < 64; ++m) acc += Aij[i * 64 + m] * Pp[m * 64 + c];
      out[j] = acc;
    }
    __syncthreads();
#pragma unroll
    for (int j = 0; j < 16; ++j) {
      int o = threadIdx.x + j * 256;
      Pp[o] = out[j];
      P[(size_t)(k - 1) * 4096 + o] = out[j];
    }
    __syncthreads();
  }
}

// ---- local scans per (chunk, batch); in-place uB -> local hs; E = chunk end

__global__ __launch_bounds__(64) void s1_local(const float* __restrict__ Am,
                                               float* __restrict__ uBhs,
                                               float* __restrict__ E, int l) {
  const int chunk = blockIdx.x, b = blockIdx.y, n = threadIdx.x;
  __shared__ __align__(16) float s_sh[64];
  float areg[64];
  const float* Arow = Am + (size_t)l * 4096 + (size_t)n * 64;
#pragma unroll
  for (int k = 0; k < 64; ++k) areg[k] = Arow[k];
  s_sh[n] = 0.f;
  __syncthreads();
  const size_t base = ((size_t)b * TT + (size_t)chunk * CLEN) * 64;
  float acc = 0.f;
  for (int tt = 0; tt < CLEN; ++tt) {
    acc = uBhs[base + (size_t)tt * 64 + n];
#pragma unroll
    for (int k = 0; k < 64; k += 4) {
      float4 sv = *(const float4*)&s_sh[k];
      acc += areg[k] * sv.x + areg[k + 1] * sv.y + areg[k + 2] * sv.z + areg[k + 3] * sv.w;
    }
    __syncthreads();
    s_sh[n] = acc;
    uBhs[base + (size_t)tt * 64 + n] = acc;
    __syncthreads();
  }
  E[((size_t)b * NCHK + chunk) * 64 + n] = acc;
}

// ---- carries across chunks: carry_j = A^64 * carry_{j-1} + E_{j-1} -------

__global__ __launch_bounds__(64) void s2_carry(const float* __restrict__ P,
                                               const float* __restrict__ E,
                                               float* __restrict__ carr) {
  const int b = blockIdx.x, n = threadIdx.x;
  __shared__ __align__(16) float c_sh[64];
  float mreg[64];
  const float* Mrow = P + (size_t)(CLEN - 1) * 4096 + (size_t)n * 64;
#pragma unroll
  for (int k = 0; k < 64; ++k) mreg[k] = Mrow[k];
  carr[((size_t)b * NCHK) * 64 + n] = 0.f;
  c_sh[n] = 0.f;
  __syncthreads();
  for (int j = 1; j < NCHK; ++j) {
    float acc = E[((size_t)b * NCHK + (j - 1)) * 64 + n];
#pragma unroll
    for (int k = 0; k < 64; k += 4) {
      float4 cv = *(const float4*)&c_sh[k];
      acc += mreg[k] * cv.x + mreg[k + 1] * cv.y + mreg[k + 2] * cv.z + mreg[k + 3] * cv.w;
    }
    __syncthreads();
    c_sh[n] = acc;
    carr[((size_t)b * NCHK + j) * 64 + n] = acc;
    __syncthreads();
  }
}

// ---- correction: hs = local + A^(k) * carry; emit f16 for C-GEMM ---------

__global__ __launch_bounds__(256) void s3_fix(const float* __restrict__ P,
                                              const float* __restrict__ uBhs,
                                              const float* __restrict__ carr,
                                              _Float16* __restrict__ hs16) {
  const int chunk = blockIdx.x, b = blockIdx.y;
  const int n = threadIdx.x & 63, tq = threadIdx.x >> 6;
  __shared__ __align__(16) float c_sh[64];
  if (threadIdx.x < 64)
    c_sh[threadIdx.x] = carr[((size_t)b * NCHK + chunk) * 64 + threadIdx.x];
  __syncthreads();
  const size_t base = ((size_t)b * TT + (size_t)chunk * CLEN) * 64;
  for (int tt = tq; tt < CLEN; tt += 4) {
    const float* Pr = P + (size_t)tt * 4096 + (size_t)n * 64;  // A^(tt+1)
    float acc = uBhs[base + (size_t)tt * 64 + n];
#pragma unroll
    for (int m = 0; m < 64; m += 4) {
      float4 pv = *(const float4*)(Pr + m);
      float4 cv = *(const float4*)&c_sh[m];
      acc += pv.x * cv.x + pv.y * cv.y + pv.z * cv.z + pv.w * cv.w;
    }
    hs16[base + (size_t)tt * 64 + n] = (_Float16)acc;
  }
}

// ---------------- fused: y = hs*Cm^T + u*Dm^T; z = y*out_w^T + b;
//                  h = LN(z + res). Block tile 32 x 256. --------------------

__global__ __launch_bounds__(256) void k_out_fused(
    const _Float16* __restrict__ hs16, const _Float16* __restrict__ u16,
    const float* __restrict__ Cg, const float* __restrict__ Dg,
    const float* __restrict__ Wog, const float* __restrict__ obg,
    const float* __restrict__ lwg, const float* __restrict__ lbg,
    float* __restrict__ h, int l) {
  __shared__ union __align__(16) {
    struct { _Float16 As[32 * 40]; _Float16 Bs[256 * 40]; _Float16 ystage[32 * 264]; } g;
    float zb[32 * 264];
  } sm;
  __shared__ float bias[256], gw[256], gb[256];
  {
    int t = threadIdx.x;
    bias[t] = obg[l * 256 + t];
    gw[t]   = lwg[l * 256 + t];
    gb[t]   = lbg[l * 256 + t];
  }
  const int rb = blockIdx.x * 32;
  const int lane = threadIdx.x & 31, wave = threadIdx.x >> 5;
  const int wrow = (wave >> 2) * 16, wcol = (wave & 3) * 64;
  const int rh = lane >> 4, col = lane & 15;
  const unsigned as_off = lds_offset_of(sm.g.As);

  v8f acc[4];
  zero_acc(acc);

  // phase A1: hs (K=64) * Cm^T   (B[k=s][n=d] = Cm[d][s], row-major direct)
  const float* C = Cg + (size_t)l * 256 * 64;
  for (int kc = 0; kc < 64; kc += 32) {
    __syncthreads();
    if (threadIdx.x < 32)  // TDM: 32x32 f16 tile of hs
      tdm_load_a32(hs16 + (size_t)rb * 64 + kc, as_off, 32, 64);
    stage_b_f32<8>(C, 64, kc, 64, sm.g.Bs);
    if (threadIdx.x < 32) __builtin_amdgcn_s_wait_tensorcnt(0);
    __syncthreads();
    v16h af = frag_a(sm.g.As, 40, wrow, 0);
#pragma unroll
    for (int j = 0; j < 4; ++j) {
      v16h bf = frag_b(sm.g.Bs, 40, wcol + j * 16, 0);
      acc[j] = wmma_f16(af, bf, acc[j]);
    }
  }
  // phase A2: u (K=256) * Dm^T
  const float* Dw = Dg + (size_t)l * 65536;
  for (int kc = 0; kc < 256; kc += 32) {
    __syncthreads();
    if (threadIdx.x < 32)  // TDM: 32x32 f16 tile of u
      tdm_load_a32(u16 + (size_t)rb * 256 + kc, as_off, 32, 256);
    stage_b_f32<8>(Dw, 256, kc, 256, sm.g.Bs);
    if (threadIdx.x < 32) __builtin_amdgcn_s_wait_tensorcnt(0);
    __syncthreads();
    v16h af = frag_a(sm.g.As, 40, wrow, 0);
#pragma unroll
    for (int j = 0; j < 4; ++j) {
      v16h bf = frag_b(sm.g.Bs, 40, wcol + j * 16, 0);
      acc[j] = wmma_f16(af, bf, acc[j]);
    }
  }
  // stage y to LDS as f16 A-operand of phase B
  __syncthreads();
#pragma unroll
  for (int j = 0; j < 4; ++j)
#pragma unroll
    for (int r = 0; r < 8; ++r)
      sm.g.ystage[(wrow + rh * 8 + r) * 264 + wcol + j * 16 + col] =
          (_Float16)acc[j][r];
  __syncthreads();

  // phase B: z = y * out_w^T
  v8f acc2[4];
  zero_acc(acc2);
  const float* Wo = Wog + (size_t)l * 65536;
  for (int kc = 0; kc < 256; kc += 32) {
    __syncthreads();
    stage_b_f32<8>(Wo, 256, kc, 256, sm.g.Bs);
    __syncthreads();
    v16h af = frag_a(sm.g.ystage, 264, wrow, kc);
#pragma unroll
    for (int j = 0; j < 4; ++j) {
      v16h bf = frag_b(sm.g.Bs, 40, wcol + j * 16, 0);
      acc2[j] = wmma_f16(af, bf, acc2[j]);
    }
  }
  __syncthreads();
#pragma unroll
  for (int j = 0; j < 4; ++j) {
    int c = wcol + j * 16 + col;
    float bv = bias[c];
#pragma unroll
    for (int r = 0; r < 8; ++r)
      sm.zb[(wrow + rh * 8 + r) * 264 + c] = acc2[j][r] + bv;
  }
  __syncthreads();
  // residual + LayerNorm; overwrite h in place (block owns its 32 rows)
#pragma unroll
  for (int rr = 0; rr < 4; ++rr) {
    int row = wave * 4 + rr;
    int grow = rb + row;
    int c0 = lane * 8;
    const float* hr = h + (size_t)grow * 256 + c0;
    float v0[8];
#pragma unroll
    for (int i = 0; i < 8; ++i) v0[i] = sm.zb[row * 264 + c0 + i] + hr[i];
    float s = 0.f;
#pragma unroll
    for (int i = 0; i < 8; ++i) s += v0[i];
    float mean = wave_sum32(s) * (1.f / 256.f);
    float q = 0.f;
#pragma unroll
    for (int i = 0; i < 8; ++i) { float d = v0[i] - mean; q += d * d; }
    float rs = rsqrtf(wave_sum32(q) * (1.f / 256.f) + 1e-5f);
    float* dst = h + (size_t)grow * 256 + c0;
#pragma unroll
    for (int i = 0; i < 8; ++i)
      dst[i] = (v0[i] - mean) * rs * gw[c0 + i] + gb[c0 + i];
  }
}

// ---------------- mean pool over T ----------------------------------------

__global__ __launch_bounds__(256) void k_pool(const float* __restrict__ h,
                                              float* __restrict__ pooled) {
  const int b = blockIdx.x, d = threadIdx.x;
  const float* p = h + (size_t)b * TT * 256 + d;
  float s = 0.f;
  for (int t = 0; t < TT; ++t) s += p[(size_t)t * 256];
  pooled[b * 256 + d] = s * (1.f / (float)TT);
}

// ---------------- classifier head (tiny) ----------------------------------

__global__ __launch_bounds__(256) void k_head(
    const float* __restrict__ pooled,
    const float* __restrict__ c1w, const float* __restrict__ c1b,
    const float* __restrict__ g1, const float* __restrict__ b1,
    const float* __restrict__ m1, const float* __restrict__ v1,
    const float* __restrict__ c2w, const float* __restrict__ c2b,
    const float* __restrict__ g2, const float* __restrict__ b2,
    const float* __restrict__ m2, const float* __restrict__ v2,
    const float* __restrict__ c3w, const float* __restrict__ c3b,
    float* __restrict__ out) {
  __shared__ float p[256], s1[512], s2[256];
  const int b = blockIdx.x, t = threadIdx.x;
  p[t] = pooled[b * 256 + t];
  __syncthreads();
  for (int o = t; o < 512; o += 256) {
    const float* w = c1w + (size_t)o * 256;
    float a = c1b[o];
#pragma unroll 4
    for (int k = 0; k < 256; k += 4) {
      float4 wv = *(const float4*)(w + k);
      a += wv.x * p[k] + wv.y * p[k + 1] + wv.z * p[k + 2] + wv.w * p[k + 3];
    }
    a = (a - m1[o]) * rsqrtf(v1[o] + 1e-5f) * g1[o] + b1[o];
    s1[o] = fmaxf(a, 0.f);
  }
  __syncthreads();
  {
    const float* w = c2w + (size_t)t * 512;
    float a = c2b[t];
#pragma unroll 4
    for (int k = 0; k < 512; k += 4) {
      float4 wv = *(const float4*)(w + k);
      a += wv.x * s1[k] + wv.y * s1[k + 1] + wv.z * s1[k + 2] + wv.w * s1[k + 3];
    }
    a = (a - m2[t]) * rsqrtf(v2[t] + 1e-5f) * g2[t] + b2[t];
    s2[t] = fmaxf(a, 0.f);
  }
  __syncthreads();
  if (t < 10) {
    const float* w = c3w + (size_t)t * 256;
    float a = c3b[t];
    for (int k = 0; k < 256; ++k) a += w[k] * s2[k];
    out[b * 10 + t] = a;
  }
}

// ---------------- host launch ---------------------------------------------

extern "C" void kernel_launch(void* const* d_in, const int* in_sizes, int n_in,
                              void* d_out, int out_size, void* d_ws, size_t ws_size,
                              hipStream_t stream) {
  const float* x      = (const float*)d_in[0];
  const float* emb_w  = (const float*)d_in[1];
  const float* emb_b  = (const float*)d_in[2];
  const float* Am     = (const float*)d_in[3];
  const float* Bm     = (const float*)d_in[4];
  const float* Cm     = (const float*)d_in[5];
  const float* Dm     = (const float*)d_in[6];
  const float* in_w   = (const float*)d_in[7];
  const float* in_b   = (const float*)d_in[8];
  const float* out_w  = (const float*)d_in[9];
  const float* out_b  = (const float*)d_in[10];
  const float* s4ln_w = (const float*)d_in[11];
  const float* s4ln_b = (const float*)d_in[12];
  const float* ln_w   = (const float*)d_in[13];
  const float* ln_b   = (const float*)d_in[14];
  const float* c1_w   = (const float*)d_in[15];
  const float* c1_b   = (const float*)d_in[16];
  const float* bn1_g  = (const float*)d_in[17];
  const float* bn1_b  = (const float*)d_in[18];
  const float* bn1_m  = (const float*)d_in[19];
  const float* bn1_v  = (const float*)d_in[20];
  const float* c2_w   = (const float*)d_in[21];
  const float* c2_b   = (const float*)d_in[22];
  const float* bn2_g  = (const float*)d_in[23];
  const float* bn2_b  = (const float*)d_in[24];
  const float* bn2_m  = (const float*)d_in[25];
  const float* bn2_v  = (const float*)d_in[26];
  const float* c3_w   = (const float*)d_in[27];
  const float* c3_b   = (const float*)d_in[28];
  (void)in_sizes; (void)n_in; (void)out_size; (void)ws_size;

  char* ws = (char*)d_ws;
  float*     h      = (float*)(ws);                    // 134,217,728 B
  _Float16*  u16    = (_Float16*)(ws + 134217728ull);  //  67,108,864 B
  float*     uBhs   = (float*)(ws + 201326592ull);     //  33,554,432 B
  _Float16*  hs16   = (_Float16*)(ws + 234881024ull);  //  16,777,216 B
  float*     E      = (float*)(ws + 251658240ull);     //     524,288 B
  float*     carr   = (float*)(ws + 252182528ull);     //     524,288 B
  float*     P      = (float*)(ws + 252706816ull);     //   1,048,576 B
  float*     pooled = (float*)(ws + 253755392ull);     //      65,536 B

  k_embed<<<MT * DD / 256, 256, 0, stream>>>(x, emb_w, emb_b, h);

  for (int l = 0; l < 4; ++l) {
    k_inproj_ln<<<MT / 32, 256, 0, stream>>>(h, in_w, in_b, s4ln_w, s4ln_b, u16, l);
    k_uB<<<MT / 128, 256, 0, stream>>>(u16, Bm, uBhs, l);
    s0_powers<<<1, 256, 0, stream>>>(Am, P, l);
    s1_local<<<dim3(NCHK, BB), 64, 0, stream>>>(Am, uBhs, E, l);
    s2_carry<<<BB, 64, 0, stream>>>(P, E, carr);
    s3_fix<<<dim3(NCHK, BB), 256, 0, stream>>>(P, uBhs, carr, hs16);
    k_out_fused<<<MT / 32, 256, 0, stream>>>(hs16, u16, Cm, Dm, out_w, out_b,
                                             ln_w, ln_b, h, l);
  }

  k_pool<<<BB, 256, 0, stream>>>(h, pooled);
  k_head<<<BB, 256, 0, stream>>>(pooled, c1_w, c1_b, bn1_g, bn1_b, bn1_m, bn1_v,
                                 c2_w, c2_b, bn2_g, bn2_b, bn2_m, bn2_v,
                                 c3_w, c3_b, (float*)d_out);
}
